// MSDeformAttnTransformerEncoderLayer_77129022701637
// MI455X (gfx1250) — matrix operand order repaired and checked by reading, not compiled, not run on hardware
//
#include <hip/hip_runtime.h>

// ---------------- problem constants ----------------
#define BQ     2
#define QLEN   13294
#define MTOT   (BQ * QLEN)        // 26588
#define DMODEL 256
#define NHEAD  8
#define NLVL   4
#define NPTS   4
#define CH     32
#define DFFN   1024

typedef __attribute__((ext_vector_type(16))) __bf16 v16bf;
typedef __attribute__((ext_vector_type(8)))  float  v8f;

// round-to-nearest-even f32 -> bf16
__device__ __forceinline__ unsigned short f2bf(float f) {
    unsigned int u = __float_as_uint(f);
    u += 0x7fffu + ((u >> 16) & 1u);
    return (unsigned short)(u >> 16);
}

union FragU {
    unsigned int u[8];
    v16bf v;
};

#define LDS_STRIDE 40   // halves per row: 32 payload + 8 pad (80B rows = 5*16B -> b128 aligned)

// ---------------- gfx1250 async global->LDS probe ----------------
#if defined(__HIP_DEVICE_COMPILE__) && __has_builtin(__builtin_amdgcn_global_load_async_to_lds_b128)
#define HAVE_ASYNC_LDS 1
#else
#define HAVE_ASYNC_LDS 0
#endif

#if HAVE_ASYNC_LDS
typedef int v4i_vec __attribute__((__vector_size__(16)));
typedef __attribute__((address_space(1))) v4i_vec* g_v4i_ptr;
typedef __attribute__((address_space(3))) v4i_vec* l_v4i_ptr;
#endif

// copy 16B (8 bf16) global -> LDS; async on CDNA5 when builtin is available
__device__ __forceinline__ void copy16_g2l(const unsigned short* __restrict__ g,
                                           unsigned short* __restrict__ l) {
#if HAVE_ASYNC_LDS
    __builtin_amdgcn_global_load_async_to_lds_b128(
        (g_v4i_ptr)(unsigned long long)(uintptr_t)g,
        (l_v4i_ptr)(unsigned int)(uintptr_t)l,
        0, 0);
#else
    *(uint4*)l = *(const uint4*)g;
#endif
}

__device__ __forceinline__ void wait_async_lds() {
#if HAVE_ASYNC_LDS
#if __has_builtin(__builtin_amdgcn_s_wait_asynccnt)
    __builtin_amdgcn_s_wait_asynccnt(0);
#else
    asm volatile("s_wait_asynccnt 0x0" ::: "memory");
#endif
#endif
}

// A fragment: 16x32 bf16 (MxK). Lanes 0-15 = M rows, half=lane>>4 selects K sub-block.
// VGPR j holds K pair: base = (j<4 ? 2j : 16+2(j-4)) + 8*half.
__device__ __forceinline__ v16bf load_a_frag(const unsigned short* As, int mrow, int lane) {
    FragU f;
    int m = mrow + (lane & 15);
    int half = lane >> 4;
    const unsigned short* base = As + m * LDS_STRIDE;
#pragma unroll
    for (int j = 0; j < 8; ++j) {
        int kb = ((j < 4) ? (2 * j) : (16 + 2 * (j - 4))) + 8 * half;
        f.u[j] = *(const unsigned int*)(base + kb);
    }
    return f.v;
}

// B fragment: 32x16 bf16 (KxN). Lane n = lane&15, VGPR j holds K pair (j<4 ? 2j : 16+2(j-4)) + 8*half.
// Bs is stored [n][k], so each lane's 16 halves are two contiguous runs of 8.
__device__ __forceinline__ v16bf load_b_frag(const unsigned short* Bs, int nbase, int lane) {
    FragU f;
    int n = nbase + (lane & 15);
    int half = lane >> 4;
    const unsigned short* p = Bs + n * LDS_STRIDE;
#pragma unroll
    for (int j = 0; j < 8; ++j) f.u[j] = *(const unsigned int*)(p + ((j < 4) ? (2 * j) : (16 + 2 * (j - 4))) + 8 * half);
    return f.v;
}

// ---------------- weight prep: f32 [K,N] -> bf16 transposed [N,K] ----------------
__global__ __launch_bounds__(256)
void convert_transpose_w(const float* __restrict__ W, unsigned short* __restrict__ Wt,
                         int K, int N)
{
    int idx = blockIdx.x * blockDim.x + threadIdx.x;
    if (idx >= K * N) return;
    int k = idx / N;
    int n = idx - k * N;                 // coalesced read over n
    Wt[(size_t)n * K + k] = f2bf(W[idx]);
}

// ---------------- tiled bf16 WMMA GEMM ----------------
// out[M,N] = (A (+A2)) [M,K] @ W [K,N] + bias[N]; W given pre-transposed bf16 Wt[N,K].
// ABF: A is bf16 [M,K] (A2/rowmask unused). OBF: write bf16 output.
// Block: 128(M) x 64(N), 256 threads = 8 waves, wave tile 32x32 (2x2 WMMA 16x16x32).
template <int ABF, int OBF>
__global__ __launch_bounds__(256)
void gemm_wmma(const void* __restrict__ Av, const float* __restrict__ A2,
               const unsigned short* __restrict__ Wt, const float* __restrict__ bias,
               const unsigned char* __restrict__ rowmask,
               void* __restrict__ outv, int N, int K, int relu)
{
    __shared__ __align__(16) unsigned short As[128 * LDS_STRIDE];
    __shared__ __align__(16) unsigned short Bs[64 * LDS_STRIDE];

    const int tid  = threadIdx.x;
    const int lane = tid & 31;
    const int wave = tid >> 5;
    const int wm   = wave >> 1;     // 0..3 (M direction)
    const int wn   = wave & 1;      // 0..1 (N direction)
    const int m0   = blockIdx.x * 128;
    const int n0   = blockIdx.y * 64;

    const float* Af = (const float*)Av;
    const unsigned short* Ab = (const unsigned short*)Av;

    v8f acc[2][2];
#pragma unroll
    for (int i = 0; i < 2; ++i)
#pragma unroll
        for (int j = 0; j < 2; ++j)
            acc[i][j] = (v8f){0.f, 0.f, 0.f, 0.f, 0.f, 0.f, 0.f, 0.f};

    for (int k0 = 0; k0 < K; k0 += 32) {
        // ---- stage A tile 128x32 ----
        if (ABF) {
            // bf16 A: pure 16B global->LDS copies (async path on CDNA5)
#pragma unroll
            for (int i = 0; i < 2; ++i) {
                int cid = tid + 256 * i;      // 0..511 chunks of 8 halves
                int row = cid >> 2;           // 0..127
                int c8  = (cid & 3) << 3;     // 0,8,16,24
                unsigned short* dst = As + row * LDS_STRIDE + c8;
                int gr = m0 + row;
                if (gr < MTOT) {
                    copy16_g2l(Ab + (size_t)gr * K + k0 + c8, dst);
                } else {
                    uint4 z = {0u, 0u, 0u, 0u};
                    *(uint4*)dst = z;
                }
            }
        } else {
            // f32 A (+optional A2): convert to bf16 while staging
#pragma unroll
            for (int i = 0; i < 4; ++i) {
                int cid = tid + 256 * i;          // 0..1023 float4 chunks
                int row = cid >> 3;               // 0..127
                int c4  = (cid & 7) << 2;         // 0,4,..,28
                float4 av = {0.f, 0.f, 0.f, 0.f};
                int gr = m0 + row;
                if (gr < MTOT) {
                    av = *(const float4*)(Af + (size_t)gr * K + k0 + c4);
                    if (A2) {
                        float4 pv = *(const float4*)(A2 + (size_t)gr * K + k0 + c4);
                        av.x += pv.x; av.y += pv.y; av.z += pv.z; av.w += pv.w;
                    }
                }
                unsigned short* d = As + row * LDS_STRIDE + c4;
                d[0] = f2bf(av.x); d[1] = f2bf(av.y); d[2] = f2bf(av.z); d[3] = f2bf(av.w);
            }
            // prefetch next A K-tile (gfx1250 global_prefetch)
            if (k0 + 32 < K) {
                int gr = m0 + (tid >> 1);
                if (gr < MTOT) __builtin_prefetch(Af + (size_t)gr * K + k0 + 32, 0, 1);
            }
        }
        // ---- stage B tile: Wt[n0..n0+63][k0..k0+31] -> Bs[n][k], 16B copies ----
        {
            int row = tid >> 2;               // n row 0..63
            int c8  = (tid & 3) << 3;         // k 0,8,16,24
            copy16_g2l(Wt + (size_t)(n0 + row) * K + k0 + c8,
                       Bs + row * LDS_STRIDE + c8);
        }
        wait_async_lds();
        __syncthreads();

        v16bf a0 = load_a_frag(As, wm * 32,      lane);
        v16bf a1 = load_a_frag(As, wm * 32 + 16, lane);
        v16bf b0 = load_b_frag(Bs, wn * 32,      lane);
        v16bf b1 = load_b_frag(Bs, wn * 32 + 16, lane);

        acc[0][0] = __builtin_amdgcn_wmma_f32_16x16x32_bf16(false, a0, false, b0, (short)0, acc[0][0], false, false);
        acc[0][1] = __builtin_amdgcn_wmma_f32_16x16x32_bf16(false, a0, false, b1, (short)0, acc[0][1], false, false);
        acc[1][0] = __builtin_amdgcn_wmma_f32_16x16x32_bf16(false, a1, false, b0, (short)0, acc[1][0], false, false);
        acc[1][1] = __builtin_amdgcn_wmma_f32_16x16x32_bf16(false, a1, false, b1, (short)0, acc[1][1], false, false);
        __syncthreads();
    }

    // ---- epilogue: C/D layout VGPR r -> (M = r + 8*half, N = lane&15) ----
    const int half = lane >> 4;
    const int lcol = lane & 15;
    float* of = (float*)outv;
    unsigned short* ob = (unsigned short*)outv;
#pragma unroll
    for (int i = 0; i < 2; ++i) {
#pragma unroll
        for (int j = 0; j < 2; ++j) {
            int n = n0 + wn * 32 + j * 16 + lcol;
            float bb = bias[n];
#pragma unroll
            for (int r = 0; r < 8; ++r) {
                int row = m0 + wm * 32 + i * 16 + r + 8 * half;
                if (row < MTOT) {
                    float val = acc[i][j][r] + bb;
                    if (relu) val = fmaxf(val, 0.f);
                    if (!ABF && rowmask && rowmask[row]) val = 0.f;
                    if (OBF) ob[(size_t)row * N + n] = f2bf(val);
                    else     of[(size_t)row * N + n] = val;
                }
            }
        }
    }
}

// ---------------- softmax over 16 (NL*NP) logits per (m, head) ----------------
__global__ __launch_bounds__(256)
void softmax16_kernel(float* __restrict__ attn)
{
    int i = blockIdx.x * blockDim.x + threadIdx.x;     // enumerates (m, h)
    if (i >= MTOT * NHEAD) return;
    float* p = attn + (size_t)i * 16;                  // m*128 + h*16 == i*16
    float4* pv = (float4*)p;
    float4 r0 = pv[0], r1 = pv[1], r2 = pv[2], r3 = pv[3];
    float v[16] = {r0.x, r0.y, r0.z, r0.w, r1.x, r1.y, r1.z, r1.w,
                   r2.x, r2.y, r2.z, r2.w, r3.x, r3.y, r3.z, r3.w};
    float mx = v[0];
#pragma unroll
    for (int k = 1; k < 16; ++k) mx = fmaxf(mx, v[k]);
    float s = 0.f;
#pragma unroll
    for (int k = 0; k < 16; ++k) { v[k] = expf(v[k] - mx); s += v[k]; }
    float inv = 1.f / s;
#pragma unroll
    for (int k = 0; k < 16; ++k) v[k] *= inv;
    pv[0] = (float4){v[0], v[1], v[2], v[3]};
    pv[1] = (float4){v[4], v[5], v[6], v[7]};
    pv[2] = (float4){v[8], v[9], v[10], v[11]};
    pv[3] = (float4){v[12], v[13], v[14], v[15]};
}

// ---------------- deformable bilinear sampling ----------------
// One wave32 per (m, head); lane = channel c. Corner fetch = coalesced 128B, L2-resident.
__global__ __launch_bounds__(256)
void deform_sample_kernel(const float* __restrict__ val, const float* __restrict__ off,
                          const float* __restrict__ attn, const float* __restrict__ refp,
                          const int* __restrict__ shapes, const int* __restrict__ lstart,
                          float* __restrict__ out)
{
    int gid  = blockIdx.x * blockDim.x + threadIdx.x;
    int wv   = gid >> 5;
    int lane = gid & 31;
    if (wv >= MTOT * NHEAD) return;
    int m = wv >> 3;
    int h = wv & 7;
    int b = (m >= QLEN) ? 1 : 0;

    const float* offp = off  + (size_t)m * 256 + h * 32;   // [l*8 + p*2 + {x,y}]
    const float* ap   = attn + (size_t)m * 128 + h * 16;   // [l*4 + p]
    const float* rp   = refp + (size_t)m * (NLVL * 2);     // [l*2 + {x,y}]
    const float* vch  = val + h * CH + lane;               // + row*256

    float acc = 0.f;
#pragma unroll
    for (int l = 0; l < NLVL; ++l) {
        int hl = shapes[2 * l], wl = shapes[2 * l + 1];
        int st = b * QLEN + lstart[l];
        float rx = rp[2 * l], ry = rp[2 * l + 1];
#pragma unroll
        for (int p = 0; p < NPTS; ++p) {
            float ox = offp[l * 8 + p * 2];
            float oy = offp[l * 8 + p * 2 + 1];
            float aw = ap[l * 4 + p];
            // loc = ref + off/(w,h); px = loc.x*w - 0.5 = rx*w + ox - 0.5
            float px = rx * (float)wl + ox - 0.5f;
            float py = ry * (float)hl + oy - 0.5f;
            float x0f = floorf(px), y0f = floorf(py);
            float fx = px - x0f, fy = py - y0f;
            int x0 = (int)x0f, y0 = (int)y0f;
            float s = 0.f;
#pragma unroll
            for (int cy = 0; cy < 2; ++cy) {
                int yi = y0 + cy;
                bool vy = (yi >= 0) && (yi < hl);
                int yc = min(max(yi, 0), hl - 1);
                float wy = cy ? fy : (1.f - fy);
#pragma unroll
                for (int cx = 0; cx < 2; ++cx) {
                    int xi = x0 + cx;
                    bool vx = (xi >= 0) && (xi < wl);
                    int xc = min(max(xi, 0), wl - 1);
                    float wx = cx ? fx : (1.f - fx);
                    float cval = vch[(size_t)(st + yc * wl + xc) * 256];
                    s += (vy && vx) ? cval * wy * wx : 0.f;
                }
            }
            acc += aw * s;
        }
    }
    out[(size_t)m * 256 + h * CH + lane] = acc;
}

// ---------------- residual + layernorm: out = LN(x + r) * g + b ----------------
// One wave32 per row (D=256 -> 8 elems/lane), wave reductions via __shfl_xor (wave32).
__global__ __launch_bounds__(256)
void ln_residual_kernel(const float* __restrict__ x, const float* __restrict__ r,
                        const float* __restrict__ g, const float* __restrict__ bta,
                        float* __restrict__ out)
{
    int gid  = blockIdx.x * blockDim.x + threadIdx.x;
    int row  = gid >> 5;
    int lane = gid & 31;
    if (row >= MTOT) return;
    const float* xr = x + (size_t)row * DMODEL;
    const float* rr = r + (size_t)row * DMODEL;
    float vals[8];
    float s = 0.f;
#pragma unroll
    for (int i = 0; i < 8; ++i) {
        int c = lane + 32 * i;
        vals[i] = xr[c] + rr[c];
        s += vals[i];
    }
#pragma unroll
    for (int o = 16; o > 0; o >>= 1) s += __shfl_xor(s, o, 32);
    float mu = s * (1.f / 256.f);
    float var = 0.f;
#pragma unroll
    for (int i = 0; i < 8; ++i) { float d = vals[i] - mu; var += d * d; }
#pragma unroll
    for (int o = 16; o > 0; o >>= 1) var += __shfl_xor(var, o, 32);
    float rs = rsqrtf(var * (1.f / 256.f) + 1e-5f);
    float* op = out + (size_t)row * DMODEL;
#pragma unroll
    for (int i = 0; i < 8; ++i) {
        int c = lane + 32 * i;
        op[c] = (vals[i] - mu) * rs * g[c] + bta[c];
    }
}

// ---------------- driver ----------------
extern "C" void kernel_launch(void* const* d_in, const int* in_sizes, int n_in,
                              void* d_out, int out_size, void* d_ws, size_t ws_size,
                              hipStream_t stream)
{
    const float* x      = (const float*)d_in[0];
    const float* pos    = (const float*)d_in[1];
    const float* refp   = (const float*)d_in[2];
    const int*   shapes = (const int*)d_in[3];
    const int*   lstart = (const int*)d_in[4];
    const unsigned char* pmask = (const unsigned char*)d_in[5];
    const float* w_off  = (const float*)d_in[6];
    const float* b_off  = (const float*)d_in[7];
    const float* w_attn = (const float*)d_in[8];
    const float* b_attn = (const float*)d_in[9];
    const float* w_val  = (const float*)d_in[10];
    const float* b_val  = (const float*)d_in[11];
    const float* w_out  = (const float*)d_in[12];
    const float* b_out  = (const float*)d_in[13];
    const float* ln1_g  = (const float*)d_in[14];
    const float* ln1_b  = (const float*)d_in[15];
    const float* w_ffn1 = (const float*)d_in[16];
    const float* b_ffn1 = (const float*)d_in[17];
    const float* w_ffn2 = (const float*)d_in[18];
    const float* b_ffn2 = (const float*)d_in[19];
    const float* ln2_g  = (const float*)d_in[20];
    const float* ln2_b  = (const float*)d_in[21];
    float* outp = (float*)d_out;

    // workspace layout (float slots): v | off | attn | samp | tmp | h | mid_bf | Wt[...]
    float* ws   = (float*)d_ws;
    float* v    = ws;
    float* off  = v    + (size_t)MTOT * 256;
    float* attn = off  + (size_t)MTOT * 256;
    float* samp = attn + (size_t)MTOT * 128;
    float* tmp  = samp + (size_t)MTOT * 256;
    float* hbuf = tmp  + (size_t)MTOT * 256;
    unsigned short* mid_bf = (unsigned short*)(hbuf + (size_t)MTOT * 256);   // M x 1024 bf16
    unsigned short* wt_val  = mid_bf + (size_t)MTOT * 1024;
    unsigned short* wt_off  = wt_val + 256 * 256;
    unsigned short* wt_attn = wt_off + 256 * 256;
    unsigned short* wt_out  = wt_attn + 256 * 128;
    unsigned short* wt_f1   = wt_out + 256 * 256;    // [1024][256]
    unsigned short* wt_f2   = wt_f1 + 256 * 1024;    // [256][1024]

    dim3 blk(256);
    int gm = (MTOT + 127) / 128;

    // --- weight prep: f32 [K,N] -> bf16 transposed [N,K] ---
    convert_transpose_w<<<(256 * 256 + 255) / 256, blk, 0, stream>>>(w_val, wt_val, 256, 256);
    convert_transpose_w<<<(256 * 256 + 255) / 256, blk, 0, stream>>>(w_off, wt_off, 256, 256);
    convert_transpose_w<<<(256 * 128 + 255) / 256, blk, 0, stream>>>(w_attn, wt_attn, 256, 128);
    convert_transpose_w<<<(256 * 256 + 255) / 256, blk, 0, stream>>>(w_out, wt_out, 256, 256);
    convert_transpose_w<<<(256 * 1024 + 255) / 256, blk, 0, stream>>>(w_ffn1, wt_f1, 256, 1024);
    convert_transpose_w<<<(1024 * 256 + 255) / 256, blk, 0, stream>>>(w_ffn2, wt_f2, 1024, 256);

    // v = x @ w_val + b_val (padding-masked rows zeroed)
    gemm_wmma<0, 0><<<dim3(gm, 256 / 64), blk, 0, stream>>>(x, nullptr, wt_val, b_val, pmask, v, 256, 256, 0);
    // off = (x + pos) @ w_off + b_off
    gemm_wmma<0, 0><<<dim3(gm, 256 / 64), blk, 0, stream>>>(x, pos, wt_off, b_off, nullptr, off, 256, 256, 0);
    // attn logits = (x + pos) @ w_attn + b_attn
    gemm_wmma<0, 0><<<dim3(gm, 128 / 64), blk, 0, stream>>>(x, pos, wt_attn, b_attn, nullptr, attn, 128, 256, 0);
    // softmax over 16 per (m, head)
    softmax16_kernel<<<(MTOT * NHEAD + 255) / 256, blk, 0, stream>>>(attn);
    // bilinear gather + weighted sum
    deform_sample_kernel<<<(MTOT * NHEAD * 32 + 255) / 256, blk, 0, stream>>>(v, off, attn, refp, shapes, lstart, samp);
    // output projection
    gemm_wmma<0, 0><<<dim3(gm, 256 / 64), blk, 0, stream>>>(samp, nullptr, wt_out, b_out, nullptr, tmp, 256, 256, 0);
    // h = LN(x + proj)
    ln_residual_kernel<<<(MTOT * 32 + 255) / 256, blk, 0, stream>>>(x, tmp, ln1_g, ln1_b, hbuf);
    // mid = relu(h @ w_ffn1 + b_ffn1), stored bf16
    gemm_wmma<0, 1><<<dim3(gm, 1024 / 64), blk, 0, stream>>>(hbuf, nullptr, wt_f1, b_ffn1, nullptr, mid_bf, 1024, 256, 1);
    // ffn2 = mid_bf16 @ w_ffn2 + b_ffn2
    gemm_wmma<1, 0><<<dim3(gm, 256 / 64), blk, 0, stream>>>(mid_bf, nullptr, wt_f2, b_ffn2, nullptr, tmp, 256, 1024, 0);
    // out = LN(h + ffn2)
    ln_residual_kernel<<<(MTOT * 32 + 255) / 256, blk, 0, stream>>>(hbuf, tmp, ln2_g, ln2_b, outp);
}
